// DAMEncoder_85306640433480
// MI455X (gfx1250) — matrix operand order repaired and checked by reading, not compiled, not run on hardware
//
#include <hip/hip_runtime.h>

// ---------------- problem constants ----------------
#define BATCH 16
#define LSEQ  1024
#define DIN   300
#define HID   256
#define RTOT  (BATCH * LSEQ)          // 16384 rows total per side

typedef __attribute__((ext_vector_type(16))) _Float16 v16h;
typedef __attribute__((ext_vector_type(8)))  _Float16 v8h;
typedef __attribute__((ext_vector_type(8)))  float    v8f;

// ---------------- WMMA helpers ----------------
__device__ __forceinline__ v16h cat8(v8h lo, v8h hi) {
  return __builtin_shufflevector(lo, hi, 0,1,2,3,4,5,6,7,8,9,10,11,12,13,14,15);
}

__device__ __forceinline__ v8f wmma16(v16h a, v16h b, v8f c) {
  return __builtin_amdgcn_wmma_f32_16x16x32_f16(false, a, false, b, (short)0, c, false, false);
}

// A operand (16x32, MxK) from row-major f16 [.,ld], tile at (m0,k0).
// lane<16: M=lane, K=k0..k0+7 | k0+16..k0+23 ; lane>=16: M=lane-16, K=k0+8.. | k0+24..
__device__ __forceinline__ v16h loadA(const _Float16* base, int ld, int m0, int k0, int lane) {
  const _Float16* p = base + (size_t)(m0 + (lane & 15)) * ld + k0 + ((lane >> 4) << 3);
  return cat8(*(const v8h*)p, *(const v8h*)(p + 16));
}

// B operand (32x16, KxN) where B[k][n] = src[n][k] (src row-major, ld = K-extent).
// lane<16: N=lane, K=k0..k0+7 | k0+16..k0+23 ; lane>=16: N=lane-16, K=k0+8.. | k0+24..
__device__ __forceinline__ v16h loadBrow(const _Float16* base, int ld, int n0, int k0, int lane) {
  const _Float16* p = base + (size_t)(n0 + (lane & 15)) * ld + k0 + ((lane >> 4) << 3);
  return cat8(*(const v8h*)p, *(const v8h*)(p + 16));
}

// B operand from pre-packed weights: [(kt*ntiles+nt)*32 + lane] * 16 halves, contiguous.
__device__ __forceinline__ v16h loadBpk(const _Float16* pk, int ntiles, int kt, int nt, int lane) {
  const _Float16* p = pk + (((size_t)kt * ntiles + nt) * 32 + lane) * 16;
  return cat8(*(const v8h*)p, *(const v8h*)(p + 8));
}

// A operand converted from fp32 row pointer (full / K-guarded tail)
__device__ __forceinline__ v16h aFromF32(const float* p, int kb) {
  v16h a;
#pragma unroll
  for (int t = 0; t < 8; ++t) { a[t] = (_Float16)p[kb + t]; a[t + 8] = (_Float16)p[kb + 16 + t]; }
  return a;
}
__device__ __forceinline__ v16h aFromF32G(const float* p, int kb, int K) {
  v16h a;
#pragma unroll
  for (int t = 0; t < 8; ++t) {
    int k1 = kb + t, k2 = kb + 16 + t;
    a[t]     = (k1 < K) ? (_Float16)p[k1] : (_Float16)0.f;
    a[t + 8] = (k2 < K) ? (_Float16)p[k2] : (_Float16)0.f;
  }
  return a;
}

// ---------------- epilogue helpers ----------------
__device__ __forceinline__ void storeE(v8f c, float bv, int rbase, int n,
                                       _Float16* __restrict__ e, _Float16* __restrict__ eT) {
#pragma unroll
  for (int r = 0; r < 8; ++r) {
    float v = c[r] + bv; v = v > 0.f ? v : 0.f;
    int m = rbase + r;
    _Float16 h = (_Float16)v;
    e[(size_t)m * HID + n] = h;
    eT[(((size_t)(m >> 10)) * HID + n) * LSEQ + (m & (LSEQ - 1))] = h;
  }
}

__device__ __forceinline__ void storeF(v8f c, float bv, int rbase, int n,
                                       const int* __restrict__ mask, _Float16* __restrict__ F) {
#pragma unroll
  for (int r = 0; r < 8; ++r) {
    float v = c[r] + bv; v = v > 0.f ? v : 0.f;
    v *= (float)mask[rbase + r];
    F[(size_t)(rbase + r) * HID + n] = (_Float16)v;
  }
}

__device__ __forceinline__ void storeV(v8f c, float bv, int rbase, int n,
                                       const int* __restrict__ mask, float* __restrict__ V) {
#pragma unroll
  for (int r = 0; r < 8; ++r) {
    float v = c[r] + bv; v = v > 0.f ? v : 0.f;
    v *= (float)mask[rbase + r];
    V[(size_t)(rbase + r) * HID + n] = v;
  }
}

// ---------------- kernels ----------------

// Pack fp32 weight [K x N] into WMMA-B layout f16, K padded to KT*32 with zeros.
__global__ void k_pack(const float* __restrict__ W, _Float16* __restrict__ dst, int K, int N) {
  int ntiles = N >> 4;
  int kt = blockIdx.x / ntiles, nt = blockIdx.x % ntiles;
  int lane = threadIdx.x;
  int n  = (nt << 4) + (lane & 15);
  int kb = (kt << 5) + ((lane >> 4) << 3);
  _Float16* p = dst + (((size_t)kt * ntiles + nt) * 32 + lane) * 16;
#pragma unroll
  for (int kk = 0; kk < 16; ++kk) {
    int k = kb + (kk & 7) + ((kk >> 3) << 4);
    p[kk] = (k < K) ? (_Float16)W[(size_t)k * N + n] : (_Float16)0.f;
  }
}

// e = relu(X @ Wp + bias), X fp32 [RTOT x 300]; 2Mx4N blocked (32x64 out/wave).
// Writes e f16 row-major [RTOT x 256] and eT f16 [BATCH][256][1024].
__global__ void k_proj(const float* __restrict__ X, const _Float16* __restrict__ Wpk,
                       const float* __restrict__ bias,
                       _Float16* __restrict__ e, _Float16* __restrict__ eT) {
  int m0 = blockIdx.x << 5, nt0 = blockIdx.y << 2, lane = threadIdx.x;
  const float* prow0 = X + (size_t)(m0 + (lane & 15)) * DIN;
  const float* prow1 = prow0 + (size_t)16 * DIN;
  v8f c[2][4] = {};
  for (int kt = 0; kt < 10; ++kt) {
    int kb = (kt << 5) + ((lane >> 4) << 3);
    v16h a[2];
    if (kt < 9) { a[0] = aFromF32(prow0, kb);       a[1] = aFromF32(prow1, kb); }
    else        { a[0] = aFromF32G(prow0, kb, DIN); a[1] = aFromF32G(prow1, kb, DIN); }
#pragma unroll
    for (int ni = 0; ni < 4; ++ni) {
      v16h b = loadBpk(Wpk, HID >> 4, kt, nt0 + ni, lane);
#pragma unroll
      for (int mi = 0; mi < 2; ++mi) c[mi][ni] = wmma16(a[mi], b, c[mi][ni]);
    }
  }
  int rb = m0 + ((lane >> 4) << 3);
#pragma unroll
  for (int ni = 0; ni < 4; ++ni) {
    int n = ((nt0 + ni) << 4) + (lane & 15);
    float bv = bias[n];
#pragma unroll
    for (int mi = 0; mi < 2; ++mi) storeE(c[mi][ni], bv, rb + (mi << 4), n, e, eT);
  }
}

// F = relu(e @ W_F + b_F) * mask  (f16 in, f16 out, row-major); 2Mx4N blocked.
__global__ void k_fgemm(const _Float16* __restrict__ e, const _Float16* __restrict__ Wpk,
                        const float* __restrict__ bias, const int* __restrict__ mask,
                        _Float16* __restrict__ F) {
  int m0 = blockIdx.x << 5, nt0 = blockIdx.y << 2, lane = threadIdx.x;
  v8f c[2][4] = {};
  for (int kt = 0; kt < 8; ++kt) {
    int k0 = kt << 5;
    v16h a[2];
    a[0] = loadA(e, HID, m0,      k0, lane);
    a[1] = loadA(e, HID, m0 + 16, k0, lane);
#pragma unroll
    for (int ni = 0; ni < 4; ++ni) {
      v16h b = loadBpk(Wpk, HID >> 4, kt, nt0 + ni, lane);
#pragma unroll
      for (int mi = 0; mi < 2; ++mi) c[mi][ni] = wmma16(a[mi], b, c[mi][ni]);
    }
  }
  int rb = m0 + ((lane >> 4) << 3);
#pragma unroll
  for (int ni = 0; ni < 4; ++ni) {
    int n = ((nt0 + ni) << 4) + (lane & 15);
    float bv = bias[n];
#pragma unroll
    for (int mi = 0; mi < 2; ++mi) storeF(c[mi][ni], bv, rb + (mi << 4), n, mask, F);
  }
}

// att[b] = Fa[b] @ Fb[b]^T  (fp32 out); 4x4 blocked (64x64 per wave).
__global__ void k_att(const _Float16* __restrict__ Fa, const _Float16* __restrict__ Fb,
                      float* __restrict__ att) {
  int b = blockIdx.z;
  const _Float16* A  = Fa + (size_t)b * LSEQ * HID;
  const _Float16* Bm = Fb + (size_t)b * LSEQ * HID;
  float* O = att + (size_t)b * LSEQ * LSEQ;
  int i0 = blockIdx.x << 6, j0 = blockIdx.y << 6, lane = threadIdx.x;
  v8f c[4][4] = {};
  for (int kt = 0; kt < 8; ++kt) {
    int k0 = kt << 5;
    v16h a[4];
#pragma unroll
    for (int mi = 0; mi < 4; ++mi) a[mi] = loadA(A, HID, i0 + (mi << 4), k0, lane);
#pragma unroll
    for (int ni = 0; ni < 4; ++ni) {
      v16h bb = loadBrow(Bm, HID, j0 + (ni << 4), k0, lane);
#pragma unroll
      for (int mi = 0; mi < 4; ++mi) c[mi][ni] = wmma16(a[mi], bb, c[mi][ni]);
    }
  }
  int j = j0 + (lane & 15);
  int rb = i0 + ((lane >> 4) << 3);
#pragma unroll
  for (int mi = 0; mi < 4; ++mi)
#pragma unroll
    for (int ni = 0; ni < 4; ++ni)
#pragma unroll
      for (int r = 0; r < 8; ++r)
        O[(size_t)(rb + (mi << 4) + r) * LSEQ + j + (ni << 4)] = c[mi][ni][r];
}

// per (b,i): rmax = max_j att, rsum = sum_j bm[j]*exp(att-rmax)
__global__ void k_rowstats(const float* __restrict__ att, const int* __restrict__ bmask,
                           float* __restrict__ rmax, float* __restrict__ rsum) {
  int bi = blockIdx.x;
  const float* row = att + (size_t)bi * LSEQ;
  const int* bm = bmask + ((bi >> 10) << 10);
  __shared__ float red[256];
  int t = threadIdx.x;
  float mx = -3.4e38f;
  for (int j = t; j < LSEQ; j += 256) mx = fmaxf(mx, row[j]);
  red[t] = mx; __syncthreads();
  for (int s = 128; s > 0; s >>= 1) { if (t < s) red[t] = fmaxf(red[t], red[t + s]); __syncthreads(); }
  mx = red[0]; __syncthreads();
  float sm = 0.f;
  for (int j = t; j < LSEQ; j += 256) sm += bm[j] ? __expf(row[j] - mx) : 0.f;
  red[t] = sm; __syncthreads();
  for (int s = 128; s > 0; s >>= 1) { if (t < s) red[t] += red[t + s]; __syncthreads(); }
  if (t == 0) { rmax[bi] = mx; rsum[bi] = red[0]; }
}

// per (b,j): cmax = max_i att, csum = sum_i am[i]*exp(att-cmax); coalesced over j
__global__ void k_colstats(const float* __restrict__ att, const int* __restrict__ amask,
                           float* __restrict__ cmax, float* __restrict__ csum) {
  int b = blockIdx.x >> 2;
  int j = ((blockIdx.x & 3) << 8) + threadIdx.x;
  const float* A = att + (size_t)b * LSEQ * LSEQ;
  const int* am = amask + (b << 10);
  float mx = -3.4e38f;
  for (int i = 0; i < LSEQ; ++i) mx = fmaxf(mx, A[(size_t)i * LSEQ + j]);
  float sm = 0.f;
  for (int i = 0; i < LSEQ; ++i) sm += am[i] ? __expf(A[(size_t)i * LSEQ + j] - mx) : 0.f;
  cmax[(b << 10) + j] = mx;
  csum[(b << 10) + j] = sm;
}

// write soft1[b][i][j] and soft2[b][j][i] as f16
__global__ void k_soft(const float* __restrict__ att, const int* __restrict__ am,
                       const int* __restrict__ bm,
                       const float* __restrict__ rmax, const float* __restrict__ rsum,
                       const float* __restrict__ cmax, const float* __restrict__ csum,
                       _Float16* __restrict__ s1, _Float16* __restrict__ s2) {
  int bi = blockIdx.x;
  int b = bi >> 10, i = bi & (LSEQ - 1);
  const float* row = att + (size_t)bi * LSEQ;
  float ami = (float)am[bi];
  float rm = rmax[bi];
  float inv1 = 1.f / (256.f * ami * rsum[bi] + 1e-8f);
  for (int j = threadIdx.x; j < LSEQ; j += 256) {
    float a = row[j];
    int bj = (b << 10) + j;
    float bmj = (float)bm[bj];
    float msk = 256.f * ami * bmj;
    s1[(size_t)bi * LSEQ + j] = (_Float16)(msk * __expf(a - rm) * inv1);
    float inv2 = 1.f / (256.f * bmj * csum[bj] + 1e-8f);
    s2[((size_t)b * LSEQ + j) * LSEQ + i] = (_Float16)(msk * __expf(a - cmax[bj]) * inv2);
  }
}

// ctx[b] = S[b] (LxL f16) @ e[b] (via eT, contiguous B fetch) -> f16 [L x H]; 4x4 blocked.
__global__ void k_ctx(const _Float16* __restrict__ S, const _Float16* __restrict__ eT,
                      _Float16* __restrict__ out) {
  int b = blockIdx.z;
  const _Float16* A  = S  + (size_t)b * LSEQ * LSEQ;
  const _Float16* BT = eT + (size_t)b * HID * LSEQ;
  int m0 = blockIdx.x << 6, n0 = blockIdx.y << 6, lane = threadIdx.x;
  v8f c[4][4] = {};
  for (int kt = 0; kt < 32; ++kt) {
    int k0 = kt << 5;
    v16h a[4];
#pragma unroll
    for (int mi = 0; mi < 4; ++mi) a[mi] = loadA(A, LSEQ, m0 + (mi << 4), k0, lane);
#pragma unroll
    for (int ni = 0; ni < 4; ++ni) {
      v16h bb = loadBrow(BT, LSEQ, n0 + (ni << 4), k0, lane);
#pragma unroll
      for (int mi = 0; mi < 4; ++mi) c[mi][ni] = wmma16(a[mi], bb, c[mi][ni]);
    }
  }
  int n = n0 + (lane & 15);
  int rb = m0 + ((lane >> 4) << 3);
#pragma unroll
  for (int mi = 0; mi < 4; ++mi)
#pragma unroll
    for (int ni = 0; ni < 4; ++ni)
#pragma unroll
      for (int r = 0; r < 8; ++r)
        out[((size_t)b * LSEQ + rb + (mi << 4) + r) * HID + n + (ni << 4)] = (_Float16)c[mi][ni][r];
}

// V = relu(concat(e, ctx) @ W_G + b_G) * mask  -> fp32 [RTOT x 256]; 2Mx4N blocked.
__global__ void k_out(const _Float16* __restrict__ e, const _Float16* __restrict__ ctx,
                      const _Float16* __restrict__ Wpk, const float* __restrict__ bias,
                      const int* __restrict__ mask, float* __restrict__ V) {
  int m0 = blockIdx.x << 5, nt0 = blockIdx.y << 2, lane = threadIdx.x;
  v8f c[2][4] = {};
  for (int kt = 0; kt < 16; ++kt) {
    v16h a[2];
    if (kt < 8) {
      a[0] = loadA(e, HID, m0,      kt << 5, lane);
      a[1] = loadA(e, HID, m0 + 16, kt << 5, lane);
    } else {
      a[0] = loadA(ctx, HID, m0,      (kt - 8) << 5, lane);
      a[1] = loadA(ctx, HID, m0 + 16, (kt - 8) << 5, lane);
    }
#pragma unroll
    for (int ni = 0; ni < 4; ++ni) {
      v16h b = loadBpk(Wpk, HID >> 4, kt, nt0 + ni, lane);
#pragma unroll
      for (int mi = 0; mi < 2; ++mi) c[mi][ni] = wmma16(a[mi], b, c[mi][ni]);
    }
  }
  int rb = m0 + ((lane >> 4) << 3);
#pragma unroll
  for (int ni = 0; ni < 4; ++ni) {
    int n = ((nt0 + ni) << 4) + (lane & 15);
    float bv = bias[n];
#pragma unroll
    for (int mi = 0; mi < 2; ++mi) storeV(c[mi][ni], bv, rb + (mi << 4), n, mask, V);
  }
}

// deterministic aggregate: per (b,h) sum_i and max_i of V
__global__ void k_agg(const float* __restrict__ V, float* __restrict__ out,
                      int sumOff, int maxOff) {
  int b = blockIdx.x, h = threadIdx.x;
  const float* P = V + (size_t)b * LSEQ * HID + h;
  float s = 0.f, mx = -3.4e38f;
  for (int i = 0; i < LSEQ; ++i) {
    float v = P[(size_t)i * HID];
    s += v; mx = fmaxf(mx, v);
  }
  out[(b << 10) + sumOff + h] = s;
  out[(b << 10) + maxOff + h] = mx;
}

// ---------------- host side ----------------
extern "C" void kernel_launch(void* const* d_in, const int* in_sizes, int n_in,
                              void* d_out, int out_size, void* d_ws, size_t ws_size,
                              hipStream_t stream) {
  const float* a_emb  = (const float*)d_in[0];
  const float* b_emb  = (const float*)d_in[1];
  const int*   a_mask = (const int*)  d_in[2];
  const int*   b_mask = (const int*)  d_in[3];
  const float* W_proj = (const float*)d_in[4];
  const float* b_proj = (const float*)d_in[5];
  const float* W_F    = (const float*)d_in[6];
  const float* b_F    = (const float*)d_in[7];
  const float* W_G    = (const float*)d_in[8];
  const float* b_G    = (const float*)d_in[9];
  float* out = (float*)d_out;

  // workspace carve-up (all sizes 256B aligned); total ~235 MB
  char* w = (char*)d_ws;
  _Float16* WpPK = (_Float16*)w; w += (size_t)10 * 16 * 32 * 16 * 2;   // 160 KB
  _Float16* WfPK = (_Float16*)w; w += (size_t)8  * 16 * 32 * 16 * 2;   // 128 KB
  _Float16* WgPK = (_Float16*)w; w += (size_t)16 * 16 * 32 * 16 * 2;   // 256 KB
  _Float16* eA   = (_Float16*)w; w += (size_t)RTOT * HID * 2;          // 8 MB
  _Float16* eB   = (_Float16*)w; w += (size_t)RTOT * HID * 2;
  _Float16* eAT  = (_Float16*)w; w += (size_t)RTOT * HID * 2;
  _Float16* eBT  = (_Float16*)w; w += (size_t)RTOT * HID * 2;
  _Float16* Fa   = (_Float16*)w; w += (size_t)RTOT * HID * 2;
  _Float16* Fb   = (_Float16*)w; w += (size_t)RTOT * HID * 2;
  _Float16* beta = (_Float16*)w; w += (size_t)RTOT * HID * 2;
  _Float16* alph = (_Float16*)w; w += (size_t)RTOT * HID * 2;
  _Float16* s1   = (_Float16*)w; w += (size_t)BATCH * LSEQ * LSEQ * 2; // 32 MB
  _Float16* s2   = (_Float16*)w; w += (size_t)BATCH * LSEQ * LSEQ * 2; // 32 MB
  float*    att  = (float*)   w; w += (size_t)BATCH * LSEQ * LSEQ * 4; // 64 MB
  float*    rmax = (float*)   w; w += (size_t)RTOT * 4;
  float*    rsum = (float*)   w; w += (size_t)RTOT * 4;
  float*    cmax = (float*)   w; w += (size_t)RTOT * 4;
  float*    csum = (float*)   w; w += (size_t)RTOT * 4;
  float*    V1   = (float*)   w; w += (size_t)RTOT * HID * 4;          // 16 MB
  float*    V2   = (float*)   w; w += (size_t)RTOT * HID * 4;          // 16 MB
  (void)ws_size; (void)in_sizes; (void)n_in; (void)out_size;

  const dim3 wave(32);
  const dim3 gRows(RTOT / 32, HID / 64);        // 512 x 4   (32x64 per wave)
  const dim3 gAtt(LSEQ / 64, LSEQ / 64, BATCH); // 16 x 16 x 16 (64x64 per wave)
  const dim3 gCtx(LSEQ / 64, HID / 64, BATCH);  // 16 x 4 x 16

  // 1. weight packing
  k_pack<<<dim3(10 * 16), wave, 0, stream>>>(W_proj, WpPK, DIN, HID);
  k_pack<<<dim3(8  * 16), wave, 0, stream>>>(W_F,    WfPK, HID, HID);
  k_pack<<<dim3(16 * 16), wave, 0, stream>>>(W_G,    WgPK, 2 * HID, HID);

  // 2. projection FFN (+ transposed copies)
  k_proj<<<gRows, wave, 0, stream>>>(a_emb, WpPK, b_proj, eA, eAT);
  k_proj<<<gRows, wave, 0, stream>>>(b_emb, WpPK, b_proj, eB, eBT);

  // 3. attend-F
  k_fgemm<<<gRows, wave, 0, stream>>>(eA, WfPK, b_F, a_mask, Fa);
  k_fgemm<<<gRows, wave, 0, stream>>>(eB, WfPK, b_F, b_mask, Fb);

  // 4. attention scores
  k_att<<<gAtt, wave, 0, stream>>>(Fa, Fb, att);

  // 5. softmax stats + masked softmax (both directions)
  k_rowstats<<<dim3(RTOT), dim3(256), 0, stream>>>(att, b_mask, rmax, rsum);
  k_colstats<<<dim3(BATCH * 4), dim3(256), 0, stream>>>(att, a_mask, cmax, csum);
  k_soft<<<dim3(RTOT), dim3(256), 0, stream>>>(att, a_mask, b_mask, rmax, rsum, cmax, csum, s1, s2);

  // 6. context vectors
  k_ctx<<<gCtx, wave, 0, stream>>>(s1, eBT, beta);   // beta  = soft1 @ b_e
  k_ctx<<<gCtx, wave, 0, stream>>>(s2, eAT, alph);   // alpha = soft2 @ a_e

  // 7. compare
  k_out<<<gRows, wave, 0, stream>>>(eA, beta, WgPK, b_G, a_mask, V1);
  k_out<<<gRows, wave, 0, stream>>>(eB, alph, WgPK, b_G, b_mask, V2);

  // 8. deterministic aggregate -> out[b][0:256]=sum v1, [256:512]=sum v2,
  //    [512:768]=max v1, [768:1024]=max v2
  k_agg<<<dim3(BATCH), dim3(HID), 0, stream>>>(V1, out, 0,       2 * HID);
  k_agg<<<dim3(BATCH), dim3(HID), 0, stream>>>(V2, out, HID,     3 * HID);
}
